// LocalTopologyBlockWithAngle_85727547228406
// MI455X (gfx1250) — matrix operand order repaired and checked by reading, not compiled, not run on hardware
//
#include <hip/hip_runtime.h>
#include <math.h>

typedef __attribute__((ext_vector_type(16))) _Float16 v16h;
typedef __attribute__((ext_vector_type(8)))  float    v8f;

#define N_NODES 8192
#define C_DIM   128
#define H_HEADS 3
#define SLOPE   0.2f
#define NCHUNK  (N_NODES / 32)     // 256 k-chunks of 32
#define CHUNK_BYTES (8 * 32 * 16 * 2)   // 8 ctiles x 32 lanes x 16 f16 = 8192 B

// ---- workspace layout (byte offsets); total = 6,553,600 bytes ----
#define WS_SU    0u                          // float[N]  : su = X @ wu
#define WS_A     (WS_SU   + N_NODES*4u)      // float[N]  : a[u] = wc*cos+ws*sin
#define WS_B     (WS_A    + N_NODES*4u)      // float[N]  : b[u] = wc*sin-ws*cos
#define WS_SCOL  (WS_B    + N_NODES*4u)      // float[N]  : column softmax sums
#define WS_CPK   (WS_SCOL + N_NODES*4u)      // float4[N] : (cos, sin, sv, m)
#define WS_SWZ   (WS_CPK  + N_NODES*16u)     // _Float16[H*N*C] swizzled Xk/colsum

__device__ __forceinline__ float* ws_f32(void* ws, unsigned off) {
    return (float*)((char*)ws + off);
}

// CDNA5 async copy global -> LDS (16B per lane), tracked by ASYNCcnt.
// GVS mode: 64-bit SGPR base + 32-bit per-lane VGPR offset; VDST = LDS byte
// offset VGPR (low 32 bits of the generic shared pointer).
__device__ __forceinline__ void async_copy_b128(unsigned lds_off, unsigned g_off,
                                                const void* sbase) {
    asm volatile("global_load_async_to_lds_b128 %0, %1, %2"
                 :: "v"(lds_off), "v"(g_off),
                    "s"((unsigned long long)(uintptr_t)sbase)
                 : "memory");
}

__device__ __forceinline__ void wait_asynccnt0() {
#if __has_builtin(__builtin_amdgcn_s_wait_asynccnt)
    __builtin_amdgcn_s_wait_asynccnt(0);
#else
    asm volatile("s_wait_asynccnt 0x0" ::: "memory");
#endif
}

// ---------------------------------------------------------------------------
// K1: per-row precompute (wave per row) + zero colsum + zero out
// ---------------------------------------------------------------------------
__global__ __launch_bounds__(256) void k1_rows(
    const float* __restrict__ X, const float* __restrict__ phis,
    const float* __restrict__ wg, void* __restrict__ ws, float* __restrict__ out)
{
    const int t    = threadIdx.x;
    const int lane = t & 31;
    const int wid  = t >> 5;
    const int row  = blockIdx.x * 8 + wid;

    // zero the (N,C) output (harness poisons it; we accumulate with atomics)
    for (int i = blockIdx.x * 256 + t; i < N_NODES * C_DIM; i += gridDim.x * 256)
        out[i] = 0.f;

    const float* xr = X + (size_t)row * C_DIM;
    float su_p = 0.f, sv_p = 0.f;
#pragma unroll
    for (int k = 0; k < 4; ++k) {
        const int c = lane + 32 * k;
        const float x = xr[c];
        su_p = fmaf(x, wg[c],          su_p);
        sv_p = fmaf(x, wg[C_DIM + c],  sv_p);
    }
#pragma unroll
    for (int off = 16; off > 0; off >>= 1) {
        su_p += __shfl_down(su_p, off);
        sv_p += __shfl_down(sv_p, off);
    }
    if (lane == 0) {
        const float wc  = wg[2 * C_DIM];
        const float wss = wg[2 * C_DIM + 1];
        const float phi = phis[row];
        const float cp = cosf(phi), sp = sinf(phi);
        ws_f32(ws, WS_SU)[row] = su_p;
        ws_f32(ws, WS_A )[row] = fmaf(wc, cp,  wss * sp);   //  wc*cos + ws*sin
        ws_f32(ws, WS_B )[row] = fmaf(wc, sp, -wss * cp);   //  wc*sin - ws*cos
        ws_f32(ws, WS_SCOL)[row] = 0.f;
        float4* cpk = (float4*)((char*)ws + WS_CPK);
        cpk[row] = make_float4(cp, sp, sv_p, 0.f);          // .w filled by k1b
    }
}

// ---------------------------------------------------------------------------
// K1b: M0 = max(su) + sqrt(wc^2+ws^2); m[v] = max(0, sv[v] + M0) -> cpk.w
// (valid softmax stabilizer since a*cos+b*sin <= sqrt(a^2+b^2) = hypot(wc,ws))
// ---------------------------------------------------------------------------
__global__ __launch_bounds__(256) void k1b_stats(
    const float* __restrict__ wg, void* __restrict__ ws)
{
    __shared__ float red[256];
    __shared__ float M0s;
    const int t = threadIdx.x;
    const float* su = ws_f32(ws, WS_SU);
    float4* cpk = (float4*)((char*)ws + WS_CPK);

    float mx = -3.4e38f;
    for (int i = t; i < N_NODES; i += 256) mx = fmaxf(mx, su[i]);
    red[t] = mx;
    __syncthreads();
    for (int s = 128; s > 0; s >>= 1) {
        if (t < s) red[t] = fmaxf(red[t], red[t + s]);
        __syncthreads();
    }
    if (t == 0) {
        const float wc = wg[2 * C_DIM], wss = wg[2 * C_DIM + 1];
        M0s = red[0] + sqrtf(wc * wc + wss * wss);
    }
    __syncthreads();
    const float M0 = M0s;
    for (int v = t; v < N_NODES; v += 256) {
        float4 p = cpk[v];
        p.w = fmaxf(0.f, p.z + M0);
        cpk[v] = p;
    }
}

// ---------------------------------------------------------------------------
// K2: colsum[v] = sum_u exp(leaky(e[u,v]) - m[v]); u-chunk staged in LDS
// ---------------------------------------------------------------------------
__global__ __launch_bounds__(256) void k2_colsum(void* __restrict__ ws)
{
    __shared__ float lsu[512], la[512], lb[512];
    const int t  = threadIdx.x;
    const int v  = blockIdx.x * 256 + t;
    const int u0 = blockIdx.y * 512;
    const float* su = ws_f32(ws, WS_SU);
    const float* aa = ws_f32(ws, WS_A);
    const float* bb = ws_f32(ws, WS_B);
    float* scol = ws_f32(ws, WS_SCOL);
    const float4* cpk = (const float4*)((const char*)ws + WS_CPK);

    for (int i = t; i < 512; i += 256) {
        lsu[i] = su[u0 + i]; la[i] = aa[u0 + i]; lb[i] = bb[u0 + i];
    }
    __syncthreads();

    const float4 p = cpk[v];
    float acc = 0.f;
    for (int i = 0; i < 512; ++i) {
        float e = fmaf(la[i], p.x, fmaf(lb[i], p.y, lsu[i] + p.z));
        e = (e >= 0.f) ? e : SLOPE * e;
        acc += __expf(e - p.w);
    }
    atomicAdd(&scol[v], acc);
}

// ---------------------------------------------------------------------------
// K3: Xk'[h,v,c] = (X[v,:] . W[h,c,:]) / colsum[v], f16, stored pre-swizzled
//     into a WMMA B-operand order with contiguous K (as A uses): 32x16 f16 tile (K=v within 32-chunk,
//     n=c within 16-tile): lane = n + 16*(K>=16), elem = K&15.
//     One (h, 32-v chunk) is exactly 8 KB contiguous -> async-copyable.
// ---------------------------------------------------------------------------
__global__ __launch_bounds__(256) void k3_xk(
    const float* __restrict__ X, const float* __restrict__ W, void* __restrict__ ws)
{
    __shared__ float Xs[64][C_DIM];
    const int t  = threadIdx.x;
    const int v0 = blockIdx.x * 64;
    const int h  = blockIdx.y;
    const float* scol = ws_f32(ws, WS_SCOL);
    _Float16* swz = (_Float16*)((char*)ws + WS_SWZ);

    for (int i = t; i < 64 * C_DIM; i += 256)
        Xs[i >> 7][i & 127] = X[(size_t)v0 * C_DIM + i];
    __syncthreads();

    const int c   = t & 127;
    const int vl0 = t >> 7;   // 0 or 1; this thread owns vl = vl0 + 2k, k=0..31
    const float* Wrow = W + ((size_t)h * C_DIM + c) * C_DIM;

    float accv[32];
#pragma unroll
    for (int k = 0; k < 32; ++k) accv[k] = 0.f;
    for (int cc = 0; cc < C_DIM; ++cc) {
        const float wv = Wrow[cc];
#pragma unroll
        for (int k = 0; k < 32; ++k)
            accv[k] = fmaf(Xs[vl0 + 2 * k][cc], wv, accv[k]);
    }
#pragma unroll
    for (int k = 0; k < 32; ++k) {
        const int v = v0 + vl0 + 2 * k;
        const float val = accv[k] / scol[v];
        const int vchunk = v >> 5, K = v & 31;
        const int ctile  = c >> 4, n = c & 15;
        const int blane  = n + ((K >> 4) << 4);
        const int e      = K & 15;
        const size_t idx = ((((size_t)h * NCHUNK + vchunk) * 8 + ctile) * 32 + blane) * 16 + e;
        swz[idx] = (_Float16)val;
    }
}

// ---------------------------------------------------------------------------
// K4: main fused attention GEMM.
//   Block = 256 threads (8 waves), one head; wave owns a 16-row u-strip x C.
//   B chunks (8 KB, identical for all 8 waves) are double-buffered in LDS via
//   CDNA5 global_load_async_to_lds_b128 (ASYNCcnt), overlapping the DMA of
//   chunk vc+1 with the exp/A-fragment math and 8 WMMAs of chunk vc.
//   B fragments are read back with ds_load_b128 at LDS latency.
// ---------------------------------------------------------------------------
union AFrag { _Float16 h[16]; v16h v; };
union BFrag { uint4 q[2];     v16h v; };

__global__ __launch_bounds__(256) void k4_attn(
    float* __restrict__ out, const void* __restrict__ ws)
{
    __shared__ __align__(16) _Float16 Bs[2][CHUNK_BYTES / 2];   // 2 x 8 KB

    const int t    = threadIdx.x;
    const int lane = t & 31;
    const int wid  = t >> 5;
    const int h    = blockIdx.y;
    const int u0   = blockIdx.x * 128 + wid * 16;

    const float* su = (const float*)((const char*)ws + WS_SU);
    const float* aa = (const float*)((const char*)ws + WS_A);
    const float* bb = (const float*)((const char*)ws + WS_B);
    const float4* cpk = (const float4*)((const char*)ws + WS_CPK);
    const char* swz = (const char*)ws + WS_SWZ;

    const int mrow = u0 + (lane & 15);
    const float su_m = su[mrow];
    const float a_m  = aa[mrow];
    const float b_m  = bb[mrow];
    const int kbase  = (lane >> 4) * 16;  // contiguous K order: upper lanes K+16

    // LDS byte offsets for the async DMA destination (low 32 bits of the
    // generic shared-memory address are the wave-relative LDS offset)
    const unsigned lds0 = (unsigned)(uintptr_t)&Bs[0][0];
    const unsigned lds1 = (unsigned)(uintptr_t)&Bs[1][0];
    const unsigned head_base = (unsigned)h * NCHUNK * CHUNK_BYTES;

    v8f acc[8];
    const v8f vzero = {};
#pragma unroll
    for (int ct = 0; ct < 8; ++ct) acc[ct] = vzero;

    // preload chunk 0 into buffer 0: 256 threads x 16B x 2 sweeps = 8 KB
#pragma unroll
    for (int i = 0; i < 2; ++i)
        async_copy_b128(lds0 + i * 4096 + t * 16,
                        head_base + i * 4096 + t * 16, swz);
    wait_asynccnt0();
    __syncthreads();

    for (int vc = 0; vc < NCHUNK; ++vc) {
        // kick off async DMA of the next chunk into the other buffer
        if (vc + 1 < NCHUNK) {
            const unsigned nxt_lds = (vc & 1) ? lds0 : lds1;
            const unsigned g0 = head_base + (unsigned)(vc + 1) * CHUNK_BYTES;
#pragma unroll
            for (int i = 0; i < 2; ++i)
                async_copy_b128(nxt_lds + i * 4096 + t * 16,
                                g0 + i * 4096 + t * 16, swz);
        }

        // build A fragment: P[m, K] = exp(leaky(e) - m[v]) as f16
        const float4* cp = cpk + vc * 32;
        AFrag af;
#pragma unroll
        for (int e = 0; e < 16; ++e) {
            const int K = kbase + e;
            const float4 p = cp[K];
            float ev = fmaf(a_m, p.x, fmaf(b_m, p.y, su_m + p.z));
            ev = (ev >= 0.f) ? ev : SLOPE * ev;
            af.h[e] = (_Float16)__expf(ev - p.w);
        }

        // preload all 8 B fragments from LDS (ds_load_b128 x16, clause-able),
        // then run the 8 WMMAs back-to-back
        const _Float16* bch = Bs[vc & 1];   // keeps addrspace(3) -> ds_load
        BFrag bf[8];
#pragma unroll
        for (int ct = 0; ct < 8; ++ct) {
            const uint4* q = (const uint4*)(bch + ((size_t)ct * 32 + lane) * 16);
            bf[ct].q[0] = q[0];
            bf[ct].q[1] = q[1];
        }
#pragma unroll
        for (int ct = 0; ct < 8; ++ct)
            acc[ct] = __builtin_amdgcn_wmma_f32_16x16x32_f16(
                false, af.v, false, bf[ct].v, (short)0, acc[ct], false, false);

        // next chunk's DMA done + all waves finished reading current buffer
        wait_asynccnt0();
        __syncthreads();
    }

    // epilogue: out[u,c] += relu(T[h,u,c])  (heads combined via atomics)
    const int rbase = (lane >> 4) * 8;
    const int ncol  = lane & 15;
#pragma unroll
    for (int ct = 0; ct < 8; ++ct) {
#pragma unroll
        for (int r = 0; r < 8; ++r) {
            const float v = fmaxf(acc[ct][r], 0.f);
            atomicAdd(&out[(size_t)(u0 + rbase + r) * C_DIM + ct * 16 + ncol], v);
        }
    }
}

// ---------------------------------------------------------------------------
extern "C" void kernel_launch(void* const* d_in, const int* in_sizes, int n_in,
                              void* d_out, int out_size, void* d_ws, size_t ws_size,
                              hipStream_t stream)
{
    const float* X    = (const float*)d_in[0];   // (N, C) f32
    const float* phis = (const float*)d_in[1];   // (N,)   f32
    const float* wg   = (const float*)d_in[2];   // (2C+2,) f32
    const float* W    = (const float*)d_in[3];   // (H, C, C) f32
    float* out = (float*)d_out;                  // (N, C) f32

    k1_rows  <<<N_NODES / 8, 256, 0, stream>>>(X, phis, wg, d_ws, out);
    k1b_stats<<<1, 256, 0, stream>>>(wg, d_ws);
    k2_colsum<<<dim3(N_NODES / 256, N_NODES / 512), 256, 0, stream>>>(d_ws);
    k3_xk    <<<dim3(N_NODES / 64, H_HEADS), 256, 0, stream>>>(X, W, d_ws);
    k4_attn  <<<dim3(N_NODES / 128, H_HEADS), 256, 0, stream>>>(out, d_ws);
}
